// FullModelSnowflake_23596550324449
// MI455X (gfx1250) — matrix operand order repaired and checked by reading, not compiled, not run on hardware
//
#include <hip/hip_runtime.h>

typedef __attribute__((ext_vector_type(16))) _Float16 v16h;
typedef __attribute__((ext_vector_type(8)))  float    v8f;
typedef _Float16 half_t;

#define LOG2E 1.44269504088896340736f
#define SCALE_ATT 0.17677669529663689f   // 1/sqrt(32)

// ---------------------------------------------------------------------------
// WMMA helpers (CDNA5 gfx1250, wave32)
// ---------------------------------------------------------------------------
__device__ __forceinline__ v8f wmma_f32_f16(v16h a, v16h b, v8f c) {
  // 8 args: (neg_a, A, neg_b, B, c_mod, C, reuse_a, reuse_b)
  return __builtin_amdgcn_wmma_f32_16x16x32_f16(false, a, false, b, (short)0, c,
                                                false, false);
}

__device__ __forceinline__ v16h ld_frag16(const half_t* p0, const half_t* p1) {
  union { v16h v; uint4 u[2]; } t;
  t.u[0] = *(const uint4*)p0;
  t.u[1] = *(const uint4*)p1;
  return t.v;
}

// A-matrix 16x32 f16 fragment. base = tile origin, ld = row stride in halves.
// Lane L holds row M=L&15; K = 16*hi .. 16*hi+15 (sequential).
__device__ __forceinline__ v16h load_a_frag(const half_t* base, int ld) {
  int lane = threadIdx.x & 31;
  int m = lane & 15, hi = lane >> 4;
  const half_t* r = base + m * ld;
  return ld_frag16(r + 16 * hi, r + 16 * hi + 8);
}

// B-matrix 32x16 f16 fragment from K-contiguous storage: base[n*ld + k].
// Lane L holds column N=L&15; K = 16*hi .. 16*hi+15 (sequential).
__device__ __forceinline__ v16h load_b_frag(const half_t* base, int ld) {
  int lane = threadIdx.x & 31;
  int n = lane & 15, hi = lane >> 4;
  const half_t* r = base + n * ld + 16 * hi;
  return ld_frag16(r, r + 8);
}

// B-matrix 32x16 f16 fragment from row-major [k][n] storage (strided reads).
// base points at the column origin; element = base[k*ld + n].
__device__ __forceinline__ v16h load_b_frag_strided(const half_t* base, int ld) {
  int lane = threadIdx.x & 31;
  int n = lane & 15, hi = lane >> 4;
  v16h r;
#pragma unroll
  for (int kk = 0; kk < 16; ++kk) r[kk] = base[(16 * hi + kk) * ld + n];
  return r;
}

// ---------------------------------------------------------------------------
// Async global -> LDS staging (CDNA5 ASYNCcnt path)
// ---------------------------------------------------------------------------
__device__ __forceinline__ void async_copy_b128(const void* gptr, void* lds) {
  unsigned lds_addr = (unsigned)(size_t)lds;  // low 32 bits = LDS byte offset
  unsigned long long ga = (unsigned long long)(size_t)gptr;
  asm volatile("global_load_async_to_lds_b128 %0, %1, off"
               :
               : "v"(lds_addr), "v"(ga)
               : "memory");
}
template <int N>
__device__ __forceinline__ void wait_async() {
#if __has_builtin(__builtin_amdgcn_s_wait_asynccnt)
  __builtin_amdgcn_s_wait_asynccnt(N);
#else
  asm volatile("s_wait_asynccnt %0" ::"i"(N) : "memory");
#endif
}

// ---------------------------------------------------------------------------
// Cross-lane reductions via ds_swizzle XOR patterns (no index VGPR needed)
// ---------------------------------------------------------------------------
template <int MASK>
__device__ __forceinline__ float swz_xor(float x) {
  return __int_as_float(
      __builtin_amdgcn_ds_swizzle(__float_as_int(x), 0x1F | (MASK << 10)));
}
__device__ __forceinline__ float rowmax16(float x) {
  x = fmaxf(x, swz_xor<1>(x));
  x = fmaxf(x, swz_xor<2>(x));
  x = fmaxf(x, swz_xor<4>(x));
  x = fmaxf(x, swz_xor<8>(x));
  return x;
}
__device__ __forceinline__ float rowsum16(float x) {
  x += swz_xor<1>(x);
  x += swz_xor<2>(x);
  x += swz_xor<4>(x);
  x += swz_xor<8>(x);
  return x;
}
__device__ __forceinline__ float wavesum32(float x) {
  x += swz_xor<1>(x);
  x += swz_xor<2>(x);
  x += swz_xor<4>(x);
  x += swz_xor<8>(x);
  x += swz_xor<16>(x);
  return x;
}

// ---------------------------------------------------------------------------
// f32 -> f16 cast
// ---------------------------------------------------------------------------
__global__ void cast_f2h(const float* __restrict__ src, half_t* __restrict__ dst,
                         int n) {
  int i = blockIdx.x * blockDim.x + threadIdx.x;
  if (i < n) dst[i] = (half_t)src[i];
}

// ---------------------------------------------------------------------------
// GEMM: C[M,N] = A[M,K](f16) * B[K,N](f16) + bias[N]
// EPI: 0 = f16 out, 1 = relu -> f16 out, 2 = f32 out
// Block: 256 thr (8 waves), tile 128x64, K-step 32. Waves 4(M) x 2(N), each
// 32x32 = 2x2 WMMA fragments. Tiles staged via async global->LDS b128 with
// double buffering: tile i+1 streams in (ASYNCcnt) while tile i is computed.
// ---------------------------------------------------------------------------
template <int EPI>
__global__ __launch_bounds__(256) void gemm_kernel(
    const half_t* __restrict__ A, const half_t* __restrict__ B,
    const float* __restrict__ bias, half_t* __restrict__ Ch,
    float* __restrict__ Cf, int M, int N, int K) {
  __shared__ __align__(16) half_t As[2][128 * 32];  // [row][k]
  __shared__ __align__(16) half_t Bs[2][32 * 64];   // [k][n] row-major

  const int m0 = blockIdx.x * 128;
  const int n0 = blockIdx.y * 64;
  const int tid = threadIdx.x;
  const int lane = tid & 31;
  const int wid = tid >> 5;
  const int wm = wid & 3;       // 0..3 -> 32-row slab
  const int wn = wid >> 2;      // 0..1 -> 32-col slab
  const int nl = lane & 15, hi = lane >> 4;

  v8f acc[2][2];
#pragma unroll
  for (int i = 0; i < 2; ++i)
#pragma unroll
    for (int j = 0; j < 2; ++j)
#pragma unroll
      for (int r = 0; r < 8; ++r) acc[i][j][r] = 0.0f;

  // cooperative-load index precompute
  const int a_row = tid >> 1;            // 0..127
  const int a_off = (tid & 1) * 16;      // 0,16
  const int b_k   = tid >> 3;            // 0..31
  const int b_no  = (tid & 7) * 8;       // 0..56

  // 3 async b128 ops per thread per tile
  auto issue_tile = [&](int k0, int buf) {
    const half_t* srcA = A + (size_t)(m0 + a_row) * K + k0 + a_off;
    async_copy_b128(srcA, &As[buf][a_row * 32 + a_off]);
    async_copy_b128(srcA + 8, &As[buf][a_row * 32 + a_off + 8]);
    const half_t* srcB = B + (size_t)(k0 + b_k) * N + n0 + b_no;
    async_copy_b128(srcB, &Bs[buf][b_k * 64 + b_no]);
  };

  const int nsteps = K >> 5;
  issue_tile(0, 0);
  for (int s = 0; s < nsteps; ++s) {
    const int buf = s & 1;
    if (s + 1 < nsteps) {
      issue_tile((s + 1) << 5, buf ^ 1);
      wait_async<3>();  // next tile's 3 ops may stay in flight
    } else {
      wait_async<0>();
    }
    __syncthreads();

    v16h a0 = load_a_frag(&As[buf][(wm * 32 + 0) * 32], 32);
    v16h a1 = load_a_frag(&As[buf][(wm * 32 + 16) * 32], 32);
    v16h b0 = load_b_frag_strided(&Bs[buf][wn * 32 + 0], 64);
    v16h b1 = load_b_frag_strided(&Bs[buf][wn * 32 + 16], 64);
    acc[0][0] = wmma_f32_f16(a0, b0, acc[0][0]);
    acc[0][1] = wmma_f32_f16(a0, b1, acc[0][1]);
    acc[1][0] = wmma_f32_f16(a1, b0, acc[1][0]);
    acc[1][1] = wmma_f32_f16(a1, b1, acc[1][1]);
    __syncthreads();  // protects buffer reuse (distance 2)
  }

  // Epilogue. C/D layout: lane holds (row = r or 8+r, col = lane&15).
#pragma unroll
  for (int fi = 0; fi < 2; ++fi) {
#pragma unroll
    for (int fj = 0; fj < 2; ++fj) {
      int col = n0 + wn * 32 + fj * 16 + nl;
      float bv = bias[col];
#pragma unroll
      for (int r = 0; r < 8; ++r) {
        int row = m0 + wm * 32 + fi * 16 + (hi ? 8 + r : r);
        float val = acc[fi][fj][r] + bv;
        if (EPI == 1) val = fmaxf(val, 0.0f);
        if (EPI == 2)
          Cf[(size_t)row * N + col] = val;
        else
          Ch[(size_t)row * N + col] = (half_t)val;
      }
    }
  }
}

// ---------------------------------------------------------------------------
// Fused flash attention with geometric bias.
// grid.x = B*H (64), grid.y = N/64 (16). 128 threads = 4 waves, 16 q-rows per
// wave. K/V 32-key tiles double-buffered in LDS via async b128: tile j+1
// streams from L2/HBM while tile j's softmax + WMMAs execute.
// ---------------------------------------------------------------------------
__global__ __launch_bounds__(128) void attn_kernel(
    const half_t* __restrict__ q, const half_t* __restrict__ k,
    const half_t* __restrict__ v, const float* __restrict__ coords,
    const float* __restrict__ alpha_p, int layer, float* __restrict__ out) {
  const int b = blockIdx.x >> 3;
  const int h = blockIdx.x & 7;
  const int qtile = blockIdx.y * 64;
  const int tid = threadIdx.x;
  const int lane = tid & 31;
  const int wid = tid >> 5;
  const int nl = lane & 15, hi = lane >> 4;

  __shared__ __align__(16) half_t Ks[2][32 * 32];  // [key][dk]
  __shared__ __align__(16) half_t Vs[2][32 * 32];  // [key][dk]
  __shared__ __align__(16) half_t Ps[4][16 * 32];  // per-wave P (16 x 32 keys)
  __shared__ float qc[64][3];

  const float al2 = alpha_p[layer] * LOG2E;
  const float sc2 = SCALE_ATT * LOG2E;

  // query-row coords for this 64-row tile
  if (tid < 192)
    qc[tid / 3][tid % 3] =
        coords[((size_t)b * 1024 + qtile + tid / 3) * 3 + tid % 3];

  // Q A-fragment, held in registers for the whole key loop
  const half_t* qrow =
      q + ((size_t)(b * 1024 + qtile + wid * 16 + nl)) * 256 + h * 32;
  v16h qa = ld_frag16(qrow + 16 * hi, qrow + 16 * hi + 8);

  v8f o0, o1;
  float mrow[8], lrow[8];
#pragma unroll
  for (int r = 0; r < 8; ++r) {
    o0[r] = 0.0f; o1[r] = 0.0f;
    mrow[r] = -3.0e38f; lrow[r] = 0.0f;
  }

  const int t_key = tid >> 2;          // 0..31
  const int t_off = (tid & 3) * 8;     // 0,8,16,24

  // 2 async b128 ops per thread per 32-key tile
  auto issue_tile = [&](int j, int buf) {
    const size_t rowg = ((size_t)(b * 1024 + j + t_key)) * 256 + h * 32 + t_off;
    async_copy_b128(k + rowg, &Ks[buf][t_key * 32 + t_off]);
    async_copy_b128(v + rowg, &Vs[buf][t_key * 32 + t_off]);
  };

  issue_tile(0, 0);
  __syncthreads();  // qc visible

  for (int jt = 0; jt < 32; ++jt) {
    const int j = jt << 5;
    const int buf = jt & 1;
    if (jt + 1 < 32) {
      issue_tile((jt + 1) << 5, buf ^ 1);
      wait_async<2>();  // next tile's 2 ops may stay in flight
    } else {
      wait_async<0>();
    }
    __syncthreads();

    v16h bk0 = load_b_frag(&Ks[buf][0], 32);       // keys j..j+15 (dk contig)
    v16h bk1 = load_b_frag(&Ks[buf][16 * 32], 32); // keys j+16..j+31
    v16h bv0 = load_b_frag_strided(&Vs[buf][0], 32);   // dk 0..15 (key strided)
    v16h bv1 = load_b_frag_strided(&Vs[buf][16], 32);  // dk 16..31

    v8f z;
#pragma unroll
    for (int r = 0; r < 8; ++r) z[r] = 0.0f;
    v8f c0 = wmma_f32_f16(qa, bk0, z);
    v8f c1 = wmma_f32_f16(qa, bk1, z);

    // geometric bias (computed in exp2 domain)
    const float* kp0 = coords + ((size_t)b * 1024 + j + nl) * 3;
    const float* kp1 = kp0 + 16 * 3;
    float k0x = kp0[0], k0y = kp0[1], k0z = kp0[2];
    float k1x = kp1[0], k1y = kp1[1], k1z = kp1[2];

    float s0[8], s1[8];
#pragma unroll
    for (int r = 0; r < 8; ++r) {
      int rm = wid * 16 + (hi ? 8 + r : r);
      float qx = qc[rm][0], qy = qc[rm][1], qz = qc[rm][2];
      s0[r] = c0[r] * sc2 + al2 * (qx * k0x + qy * k0y + qz * k0z);
      s1[r] = c1[r] * sc2 + al2 * (qx * k1x + qy * k1y + qz * k1z);
    }

    // online softmax: per-row stats replicated across each 16-lane half
#pragma unroll
    for (int r = 0; r < 8; ++r) {
      float mx = rowmax16(fmaxf(s0[r], s1[r]));
      float mn = fmaxf(mrow[r], mx);
      float corr = exp2f(mrow[r] - mn);
      float p0 = exp2f(s0[r] - mn);
      float p1 = exp2f(s1[r] - mn);
      lrow[r] = lrow[r] * corr + rowsum16(p0 + p1);
      mrow[r] = mn;
      o0[r] *= corr;
      o1[r] *= corr;
      int rm = hi ? 8 + r : r;
      Ps[wid][rm * 32 + nl] = (half_t)p0;
      Ps[wid][rm * 32 + 16 + nl] = (half_t)p1;
    }
    __builtin_amdgcn_wave_barrier();  // wave-internal LDS transpose ordering

    v16h pa = load_a_frag(&Ps[wid][0], 32);
    o0 = wmma_f32_f16(pa, bv0, o0);
    o1 = wmma_f32_f16(pa, bv1, o1);
    __syncthreads();  // protects Ks/Vs buffer reuse (distance 2)
  }

#pragma unroll
  for (int r = 0; r < 8; ++r) {
    float inv = 1.0f / lrow[r];
    int rowg = b * 1024 + qtile + wid * 16 + (hi ? 8 + r : r);
    out[(size_t)rowg * 256 + h * 32 + nl] = o0[r] * inv;
    out[(size_t)rowg * 256 + h * 32 + 16 + nl] = o1[r] * inv;
  }
}

// ---------------------------------------------------------------------------
// residual + LayerNorm; one block per token row, writes f32 state + f16 copy
// ---------------------------------------------------------------------------
__global__ __launch_bounds__(256) void resid_ln_kernel(
    const float* __restrict__ a, const float* __restrict__ res,
    const float* __restrict__ g, const float* __restrict__ be,
    float* __restrict__ xout, half_t* __restrict__ xh) {
  const int row = blockIdx.x;
  const int t = threadIdx.x;
  const size_t idx = (size_t)row * 256 + t;
  float val = a[idx] + res[idx];

  __shared__ float red[8];
  float s = wavesum32(val);
  if ((t & 31) == 0) red[t >> 5] = s;
  __syncthreads();
  float tot = 0.0f;
#pragma unroll
  for (int i = 0; i < 8; ++i) tot += red[i];
  const float mu = tot * (1.0f / 256.0f);
  __syncthreads();

  float d = val - mu;
  s = wavesum32(d * d);
  if ((t & 31) == 0) red[t >> 5] = s;
  __syncthreads();
  tot = 0.0f;
#pragma unroll
  for (int i = 0; i < 8; ++i) tot += red[i];
  const float var = tot * (1.0f / 256.0f);

  float y = d * rsqrtf(var + 1e-5f) * g[t] + be[t];
  xout[idx] = y;
  xh[idx] = (half_t)y;
}

// ---------------------------------------------------------------------------
// host: orchestrate 4 layers
// ---------------------------------------------------------------------------
extern "C" void kernel_launch(void* const* d_in, const int* in_sizes, int n_in,
                              void* d_out, int out_size, void* d_ws,
                              size_t ws_size, hipStream_t stream) {
  (void)in_sizes; (void)n_in; (void)out_size; (void)ws_size;
  const float* x      = (const float*)d_in[0];
  const float* coords = (const float*)d_in[1];
  const float* Wq = (const float*)d_in[2];  const float* bq = (const float*)d_in[3];
  const float* Wk = (const float*)d_in[4];  const float* bk = (const float*)d_in[5];
  const float* Wv = (const float*)d_in[6];  const float* bv = (const float*)d_in[7];
  const float* alpha = (const float*)d_in[8];
  const float* W1 = (const float*)d_in[9];  const float* b1 = (const float*)d_in[10];
  const float* W2 = (const float*)d_in[11]; const float* b2 = (const float*)d_in[12];
  const float* g1 = (const float*)d_in[13]; const float* be1 = (const float*)d_in[14];
  const float* g2 = (const float*)d_in[15]; const float* be2 = (const float*)d_in[16];

  constexpr int Ll = 4, Dd = 256, FFf = 1024;
  const int T = 8 * 1024;  // tokens

  char* p = (char*)d_ws;
  auto carve = [&](size_t bytes) -> char* {
    char* r = p;
    p += (bytes + 255) & ~(size_t)255;
    return r;
  };
  half_t* wq_h = (half_t*)carve((size_t)Ll * Dd * Dd * 2);
  half_t* wk_h = (half_t*)carve((size_t)Ll * Dd * Dd * 2);
  half_t* wv_h = (half_t*)carve((size_t)Ll * Dd * Dd * 2);
  half_t* w1_h = (half_t*)carve((size_t)Ll * Dd * FFf * 2);
  half_t* w2_h = (half_t*)carve((size_t)Ll * FFf * Dd * 2);
  half_t* xh   = (half_t*)carve((size_t)T * Dd * 2);
  half_t* qh   = (half_t*)carve((size_t)T * Dd * 2);
  half_t* kh   = (half_t*)carve((size_t)T * Dd * 2);
  half_t* vh   = (half_t*)carve((size_t)T * Dd * 2);
  half_t* ffh  = (half_t*)carve((size_t)T * FFf * 2);
  float*  tmp  = (float*)carve((size_t)T * Dd * 4);
  float*  xbuf = (float*)carve((size_t)T * Dd * 4);

  auto cast = [&](const float* src, half_t* dst, int n) {
    cast_f2h<<<(n + 255) / 256, 256, 0, stream>>>(src, dst, n);
  };
  cast(Wq, wq_h, Ll * Dd * Dd);
  cast(Wk, wk_h, Ll * Dd * Dd);
  cast(Wv, wv_h, Ll * Dd * Dd);
  cast(W1, w1_h, Ll * Dd * FFf);
  cast(W2, w2_h, Ll * FFf * Dd);
  cast(x, xh, T * Dd);

  const float* cur_x = x;
  const dim3 gD(T / 128, Dd / 64);    // N = 256
  const dim3 gF(T / 128, FFf / 64);   // N = 1024
  const dim3 gA(64, 1024 / 64);       // (B*H, q-tiles)

  for (int l = 0; l < Ll; ++l) {
    gemm_kernel<0><<<gD, 256, 0, stream>>>(xh, wq_h + (size_t)l * Dd * Dd,
                                           bq + l * Dd, qh, nullptr, T, Dd, Dd);
    gemm_kernel<0><<<gD, 256, 0, stream>>>(xh, wk_h + (size_t)l * Dd * Dd,
                                           bk + l * Dd, kh, nullptr, T, Dd, Dd);
    gemm_kernel<0><<<gD, 256, 0, stream>>>(xh, wv_h + (size_t)l * Dd * Dd,
                                           bv + l * Dd, vh, nullptr, T, Dd, Dd);
    attn_kernel<<<gA, 128, 0, stream>>>(qh, kh, vh, coords, alpha, l, tmp);
    resid_ln_kernel<<<T, 256, 0, stream>>>(tmp, cur_x, g1 + l * Dd,
                                           be1 + l * Dd, xbuf, xh);
    cur_x = xbuf;
    gemm_kernel<1><<<gF, 256, 0, stream>>>(xh, w1_h + (size_t)l * Dd * FFf,
                                           b1 + l * FFf, ffh, nullptr, T, FFf, Dd);
    gemm_kernel<2><<<gD, 256, 0, stream>>>(ffh, w2_h + (size_t)l * FFf * Dd,
                                           b2 + l * Dd, nullptr, tmp, T, Dd, FFf);
    float* xdst = (l == Ll - 1) ? (float*)d_out : xbuf;
    resid_ln_kernel<<<T, 256, 0, stream>>>(tmp, cur_x, g2 + l * Dd,
                                           be2 + l * Dd, xdst, xh);
    cur_x = xdst;
  }
}
